// FastVectorLIFCell_65171833750166
// MI455X (gfx1250) — matrix operand order repaired and checked by reading, not compiled, not run on hardware
//
#include <hip/hip_runtime.h>

// ---------------------------------------------------------------------------
// FastVectorLIFCell on MI455X (gfx1250, wave32)
//   pass 1: f32 -> bf16 conversion of X and W (bandwidth-bound)
//   pass 2: cur = X @ W^T  bf16 WMMA GEMM with double-buffered
//           global_load_async_to_lds_b128 staging (ASYNCcnt-tracked)
//   pass 3: LIF scan over T (bandwidth-bound chain kernel)
// ---------------------------------------------------------------------------

typedef __attribute__((ext_vector_type(16))) __bf16 v16bf;
typedef __attribute__((ext_vector_type(8)))  __bf16 v8bf;
typedef __attribute__((ext_vector_type(4)))  __bf16 v4bf;
typedef __attribute__((ext_vector_type(8)))  float  v8f;

#define B_DIM 64
#define T_DIM 512
#define D_DIM 1024
#define H_DIM 1024
#define M_DIM (B_DIM * T_DIM)      // 32768 GEMM rows

#define BM 128
#define BN 128
#define BK 32
#define LDS_STRIDE 40              // bf16 elems per LDS row: 32 data + 8 pad = 80B
#define LDS_ROW_BYTES 80           // stride 80B -> conflict-free 16-row frag reads
#define BUF_BYTES (BM * LDS_ROW_BYTES)   // 10240 B per tile buffer

// ---------------------------------------------------------------------------
// f32 -> bf16 streaming conversion (4 elements / thread)
// ---------------------------------------------------------------------------
__global__ __launch_bounds__(256)
void cvt_f32_bf16(const float* __restrict__ src, __bf16* __restrict__ dst)
{
    const size_t i4 = (size_t)blockIdx.x * 256 + threadIdx.x;
    const float4 v = *(const float4*)(src + i4 * 4);
    v4bf p;
    p[0] = (__bf16)v.x; p[1] = (__bf16)v.y;
    p[2] = (__bf16)v.z; p[3] = (__bf16)v.w;
    *(v4bf*)(dst + i4 * 4) = p;
}

// ---------------------------------------------------------------------------
// bf16 WMMA GEMM: Cur[m][n] = sum_k Xb[m][k] * Wb[n][k]
// 256 threads = 8 wave32s; block tile 128x128, K-tile 32; wave tile 64x32.
// LDS tiles double-buffered, filled by global_load_async_to_lds_b128.
// ---------------------------------------------------------------------------
__global__ __launch_bounds__(256)
void lif_gemm_bf16_wmma(const __bf16* __restrict__ Xb,
                        const __bf16* __restrict__ Wb,
                        float* __restrict__ Cur)
{
    __shared__ __bf16 As[2][BM * LDS_STRIDE];   // [m][k] tiles of X
    __shared__ __bf16 Bs[2][BN * LDS_STRIDE];   // [n][k] tiles of W

    const int tid  = threadIdx.x;
    const int lane = tid & 31;
    const int wid  = tid >> 5;                  // 8 waves
    const int wm   = (wid >> 2) * 64;           // wave M offset (0,64)
    const int wn   = (wid & 3)  * 32;           // wave N offset (0..96)
    const int m0   = blockIdx.y * BM;
    const int n0   = blockIdx.x * BN;

    const int halfsel = (lane < 16) ? 0 : 1;
    const int lr      = lane & 15;

    // LDS byte offsets (flat shared addr truncated to 32b == LDS offset)
    const unsigned asBase = (unsigned)(uintptr_t)&As[0][0];
    const unsigned bsBase = (unsigned)(uintptr_t)&Bs[0][0];

    // per-thread staging chunks: 512 16B-chunks per tile, 2 per thread per matrix
    const int ch0  = tid;              // chunk ids: tid, tid+256
    const int row0 = ch0 >> 2,  c0 = ch0 & 3;
    const int ch1  = tid + 256;
    const int row1 = ch1 >> 2,  c1 = ch1 & 3;

    // issue one tile-set (4 async b128 loads / thread) into buffer `buf`
    auto stage = [&](int buf, int kt) {
        const unsigned aOff = asBase + (unsigned)buf * BUF_BYTES;
        const unsigned bOff = bsBase + (unsigned)buf * BUF_BYTES;
        {
            unsigned la = aOff + row0 * LDS_ROW_BYTES + c0 * 16;
            const __bf16* ga = Xb + (size_t)(m0 + row0) * D_DIM + kt + c0 * 8;
            asm volatile("global_load_async_to_lds_b128 %0, %1, off"
                         :: "v"(la), "v"(ga) : "memory");
            unsigned lb = bOff + row0 * LDS_ROW_BYTES + c0 * 16;
            const __bf16* gb = Wb + (size_t)(n0 + row0) * D_DIM + kt + c0 * 8;
            asm volatile("global_load_async_to_lds_b128 %0, %1, off"
                         :: "v"(lb), "v"(gb) : "memory");
        }
        {
            unsigned la = aOff + row1 * LDS_ROW_BYTES + c1 * 16;
            const __bf16* ga = Xb + (size_t)(m0 + row1) * D_DIM + kt + c1 * 8;
            asm volatile("global_load_async_to_lds_b128 %0, %1, off"
                         :: "v"(la), "v"(ga) : "memory");
            unsigned lb = bOff + row1 * LDS_ROW_BYTES + c1 * 16;
            const __bf16* gb = Wb + (size_t)(n0 + row1) * D_DIM + kt + c1 * 8;
            asm volatile("global_load_async_to_lds_b128 %0, %1, off"
                         :: "v"(lb), "v"(gb) : "memory");
        }
    };

    v8f acc[4][2];
#pragma unroll
    for (int i = 0; i < 4; ++i)
#pragma unroll
        for (int j = 0; j < 2; ++j)
#pragma unroll
            for (int r = 0; r < 8; ++r)
                acc[i][j][r] = 0.0f;

    constexpr int NS = D_DIM / BK;   // 32 K-steps
    stage(0, 0);                     // prologue: 4 outstanding async loads

    for (int s = 0; s < NS; ++s) {
        const int buf = s & 1;
        if (s + 1 < NS) {
            stage(buf ^ 1, (s + 1) * BK);                 // 8 outstanding
            asm volatile("s_wait_asynccnt 0x4" ::: "memory");  // current buf done
        } else {
            asm volatile("s_wait_asynccnt 0x0" ::: "memory");
        }
        __syncthreads();   // all waves' current-buffer data visible

        // ---- A fragments (16x32 bf16): lanes 0-15: M=lane, K{0-7,16-23};
        //      lanes 16-31: M=lane-16, K{8-15,24-31}
        v16bf afr[4];
#pragma unroll
        for (int i = 0; i < 4; ++i) {
            const __bf16* base = &As[buf][(wm + 16 * i + lr) * LDS_STRIDE];
            v8bf lo = *(const v8bf*)(base + halfsel * 8);
            v8bf hi = *(const v8bf*)(base + 16 + halfsel * 8);
            afr[i] = __builtin_shufflevector(lo, hi,
                0,1,2,3,4,5,6,7,8,9,10,11,12,13,14,15);
        }
        // ---- B fragments (32x16, K x N): lanes 0-15: N=lane, K{0-7,16-23};
        //      lanes 16-31: N=lane-16, K{8-15,24-31}
        v16bf bfr[2];
#pragma unroll
        for (int j = 0; j < 2; ++j) {
            const __bf16* base = &Bs[buf][(wn + 16 * j + lr) * LDS_STRIDE];
            v8bf lo = *(const v8bf*)(base + halfsel * 8);
            v8bf hi = *(const v8bf*)(base + 16 + halfsel * 8);
            bfr[j] = __builtin_shufflevector(lo, hi,
                0,1,2,3,4,5,6,7,8,9,10,11,12,13,14,15);
        }
        // ---- 8 WMMAs per K-step
#pragma unroll
        for (int i = 0; i < 4; ++i)
#pragma unroll
            for (int j = 0; j < 2; ++j)
                acc[i][j] = __builtin_amdgcn_wmma_f32_16x16x32_bf16(
                    false, afr[i], false, bfr[j],
                    (short)0, acc[i][j], false, false);

        __syncthreads();   // buffer fully consumed before it is refilled at s+1
    }

    // ---- store C: lanes 0-15 -> M rows r, lanes 16-31 -> M rows r+8
    const int mhalf = halfsel * 8;
#pragma unroll
    for (int i = 0; i < 4; ++i) {
#pragma unroll
        for (int j = 0; j < 2; ++j) {
            const int gm = m0 + wm + 16 * i + mhalf;
            const int gn = n0 + wn + 16 * j + lr;
#pragma unroll
            for (int r = 0; r < 8; ++r)
                Cur[(size_t)(gm + r) * H_DIM + gn] = acc[i][j][r];
        }
    }
}

// ---------------------------------------------------------------------------
// LIF scan: one thread per (b, h) chain; coalesced across h within a wave.
// ---------------------------------------------------------------------------
__global__ __launch_bounds__(256)
void lif_scan(const float* __restrict__ Cur, float* __restrict__ Out)
{
    const int idx = blockIdx.x * 256 + threadIdx.x;   // 0 .. B*H-1
    const int b   = idx >> 10;                        // / H_DIM
    const int h   = idx & (H_DIM - 1);
    const float* c = Cur + (size_t)b * T_DIM * H_DIM + h;
    float*       o = Out + (size_t)b * T_DIM * H_DIM + h;

    float m = 0.0f;
    for (int t = 0; t < T_DIM; ++t) {
        const float v = 0.9f * m + c[(size_t)t * H_DIM];   // LIF_DECAY*m + cur
        const bool  s = (v > 1.0f);                         // threshold
        m = s ? 0.9f * v : v;                               // v(1-z) + ALPHA*v*z
        o[(size_t)t * H_DIM] = s ? 1.0f : 0.0f;
    }
}

extern "C" void kernel_launch(void* const* d_in, const int* in_sizes, int n_in,
                              void* d_out, int out_size, void* d_ws, size_t ws_size,
                              hipStream_t stream)
{
    const float* X  = (const float*)d_in[0];   // [B, T, D] f32
    const float* Wm = (const float*)d_in[1];   // [H, D]    f32
    float* out = (float*)d_out;                // [B, T, H] f32 spikes

    // workspace layout (all offsets 2MB-aligned):
    //   Xb : bf16 [M, D]  64 MB @ 0
    //   Wb : bf16 [H, D]   2 MB @ 64 MB
    //   cur: f32  [M, H] 128 MB @ 66 MB
    char* ws = (char*)d_ws;
    __bf16* Xb  = (__bf16*)(ws);
    __bf16* Wb  = (__bf16*)(ws + (size_t)M_DIM * D_DIM * 2);
    float*  cur = (float*)(ws + (size_t)M_DIM * D_DIM * 2 + (size_t)H_DIM * D_DIM * 2);

    // pass 1: precision conversion (streams once; also halves GEMM read traffic)
    const size_t nx4 = (size_t)M_DIM * D_DIM / 4;   // 8388608 float4s
    cvt_f32_bf16<<<(unsigned)(nx4 / 256), 256, 0, stream>>>(X, Xb);
    const size_t nw4 = (size_t)H_DIM * D_DIM / 4;   // 262144 float4s
    cvt_f32_bf16<<<(unsigned)(nw4 / 256), 256, 0, stream>>>(Wm, Wb);

    // pass 2: WMMA GEMM
    dim3 grid_gemm(H_DIM / BN, M_DIM / BM);         // (8, 256)
    lif_gemm_bf16_wmma<<<grid_gemm, 256, 0, stream>>>(Xb, Wb, cur);

    // pass 3: temporal scan
    const int nchain = B_DIM * H_DIM;               // 65536 chains
    lif_scan<<<nchain / 256, 256, 0, stream>>>(cur, out);
}
